// HPNActor_1546188226865
// MI455X (gfx1250) — matrix operand order repaired and checked by reading, not compiled, not run on hardware
//
#include <hip/hip_runtime.h>
#include <hip/hip_bf16.h>
#include <math.h>

// HPN actor critic collapsed to one GEMM:  tmp[16384, 256] = G[16384, 848] @ W2stack[848, 256]
// (G = entity-summed outer(h,[feats,1]) per branch + bias coeffs; W2stack = rearranged W2+b2.)
// GEMM runs as split-precision f16 WMMA (v_wmma_f32_16x16x32_f16, 8192 MAC/instr):
//   a*b ~= a_hi*b_hi + a_hi*b_lo + a_lo*b_hi   (f16 hi + f16 residual ~ 22 mantissa bits,
//   f32 accumulate in the WMMA) -> ~fp32 accuracy at ~2.6x fewer matrix issue slots than
//   the 16x16x4 fp32 path. K padded 848 -> 864 (27 blocks of 32).

#define BSZ      16384
#define KTOT     864           // padded K (multiple of 32)
#define KS       872           // G row stride in halves (1744 B, 16B-aligned, bank-spread)
#define KPS      872           // packed W2 row stride in halves
#define MT       32            // samples per workgroup (two 16-row WMMA tiles)

typedef __attribute__((ext_vector_type(2)))  float    v2f;
typedef __attribute__((ext_vector_type(8)))  float    v8f;
typedef __attribute__((ext_vector_type(8)))  _Float16 h8;
typedef __attribute__((ext_vector_type(16))) _Float16 v16h;

// A fragment (16x32 f16): lane needs 8 halves at p and 8 at p+16 (K split per ISA layout).
__device__ __forceinline__ v16h ldA(const _Float16* p) {
    union { v16h v; h8 h[2]; } u;
    u.h[0] = *(const h8*)(p);
    u.h[1] = *(const h8*)(p + 16);
    return u.v;
}
// B fragment (32x16 f16): lane needs 8 halves at p and 8 at p+16 (K-major storage, same K split as A).
__device__ __forceinline__ v16h ldB(const _Float16* p) {
    union { v16h v; h8 h[2]; } u;
    u.h[0] = *(const h8*)(p);
    u.h[1] = *(const h8*)(p + 16);
    return u.v;
}
// Split a float into f16 hi + f16 residual, store to hi/lo arrays.
__device__ __forceinline__ void putHL(_Float16* Hh, _Float16* Hl, int idx, float g) {
    _Float16 h = (_Float16)g;
    Hh[idx] = h;
    Hl[idx] = (_Float16)(g - (float)h);
}

// ---------------------------------------------------------------------------
// Kernel 0: pack W2stack (K-major, hi/lo f16) into workspace.
//   K rows   0..319 : ally  W2[j, k*256+n], row = j*5+k
//        320..639   : adv   W2
//        640..831   : lm    W2 (j*3+k)
//        832..836   : ally b2 ; 837..841: adv b2 ; 842..844: lm b2 ; 845..871: zero
// Output: W2h[n*KPS + k], W2l[n*KPS + k]   (n = 0..255)
// ---------------------------------------------------------------------------
__global__ __launch_bounds__(256) void pack_w2(
    const float* __restrict__ aW2, const float* __restrict__ ab2,
    const float* __restrict__ dW2, const float* __restrict__ db2,
    const float* __restrict__ lW2, const float* __restrict__ lb2,
    _Float16* __restrict__ W2h, _Float16* __restrict__ W2l)
{
    int n = blockIdx.x;
    #pragma unroll
    for (int it = 0; it < 4; ++it) {
        int k = threadIdx.x + it * 256;
        if (k >= KPS) break;
        float v = 0.0f;
        if (k < 320) {
            int j = k / 5, kk = k % 5;
            v = aW2[j * 1280 + kk * 256 + n];
        } else if (k < 640) {
            int j = (k - 320) / 5, kk = (k - 320) % 5;
            v = dW2[j * 1280 + kk * 256 + n];
        } else if (k < 832) {
            int j = (k - 640) / 3, kk = (k - 640) % 3;
            v = lW2[j * 768 + kk * 256 + n];
        } else if (k < 837) {
            v = ab2[(k - 832) * 256 + n];
        } else if (k < 842) {
            v = db2[(k - 837) * 256 + n];
        } else if (k < 845) {
            v = lb2[(k - 842) * 256 + n];
        }
        _Float16 h = (_Float16)v;
        W2h[n * KPS + k] = h;
        W2l[n * KPS + k] = (_Float16)(v - (float)h);
    }
}

// ---------------------------------------------------------------------------
// Kernel 1: fused prep (hi/lo f16 G build) + split-f16 WMMA GEMM + epilogue.
// Block = 256 threads (8 wave32), MT=32 samples; each wave owns 2x2 16x16 C tiles.
// ---------------------------------------------------------------------------
__global__ __launch_bounds__(256) void hpn_main(
    const float* __restrict__ inp,   // [BS,30]
    const float* __restrict__ act,   // [BS,2]
    const float* __restrict__ selfW, const float* __restrict__ selfb,   // 4x64, 64
    const float* __restrict__ aW1,   const float* __restrict__ ab1,     // 4x64, 64
    const float* __restrict__ dW1,   const float* __restrict__ db1,     // 4x64, 64
    const float* __restrict__ lW1,   const float* __restrict__ lb1,     // 2x64, 64
    const float* __restrict__ mw,    // merger_w [1,4,64]
    const float* __restrict__ f1W,   const float* __restrict__ f1b,     // 66x64, 64
    const float* __restrict__ f2W,   const float* __restrict__ f2b,     // 64x1, 1
    const _Float16* __restrict__ W2h, const _Float16* __restrict__ W2l, // packed [256][KPS]
    float* __restrict__ outq)        // [BS] q  ++  [BS*64] x
{
    // Overlaid LDS: phase 1/2 use Ghi/Glo [MT][KS] f16; epilogue reuses the bytes as floats.
    __shared__ __align__(16) float smem[MT * KS / 2 * 2];  // 27904 floats = 111,616 B
    _Float16* Ghi = (_Float16*)smem;           // [MT][KS]
    _Float16* Glo = Ghi + MT * KS;             // [MT][KS]
    float* tmpS = smem;                        // [MT][260]
    float* o66S = smem + MT * 260;             // [MT][68]
    float* xS   = o66S + MT * 68;              // [MT][64]
    float* swS  = xS + MT * 64;                // [4][64] softmax(merger_w)

    const int t  = threadIdx.x;
    const int s0 = blockIdx.x * MT;

    // ------------------ Phase 1: build hi/lo G tile in LDS ------------------
    // item = (sample s, hyper-hidden j); 32*64 = 2048 items over 256 threads.
    #pragma unroll
    for (int it = 0; it < 8; ++it) {
        int item = t + it * 256;
        int s = item >> 6;
        int j = item & 63;
        const float* in = inp + (size_t)(s0 + s) * 30;
        _Float16* Hh = Ghi + s * KS;
        _Float16* Hl = Glo + s * KS;
        // ally branch (d=4, 3 entities): pos cols 10..15, vel cols 20..25
        {
            float w0 = aW1[j], w1 = aW1[64 + j], w2 = aW1[128 + j], w3 = aW1[192 + j];
            float b = ab1[j];
            float g0 = 0.f, g1 = 0.f, g2 = 0.f, g3 = 0.f, g4 = 0.f;
            #pragma unroll
            for (int n = 0; n < 3; ++n) {
                float f0 = in[10 + 2 * n], f1 = in[11 + 2 * n];
                float f2 = in[20 + 2 * n], f3 = in[21 + 2 * n];
                float p = f0 * w0 + f1 * w1 + f2 * w2 + f3 * w3 + b;
                float h = p > 0.f ? p : 0.01f * p;   // leaky_relu(0.01)
                g0 += h * f0; g1 += h * f1; g2 += h * f2; g3 += h * f3; g4 += h;
            }
            int o = j * 5;
            putHL(Hh, Hl, o + 0, g0); putHL(Hh, Hl, o + 1, g1);
            putHL(Hh, Hl, o + 2, g2); putHL(Hh, Hl, o + 3, g3);
            putHL(Hh, Hl, o + 4, g4);
        }
        // adversary branch (d=4, 2 entities): pos cols 16..19, vel cols 26..29
        {
            float w0 = dW1[j], w1 = dW1[64 + j], w2 = dW1[128 + j], w3 = dW1[192 + j];
            float b = db1[j];
            float g0 = 0.f, g1 = 0.f, g2 = 0.f, g3 = 0.f, g4 = 0.f;
            #pragma unroll
            for (int n = 0; n < 2; ++n) {
                float f0 = in[16 + 2 * n], f1 = in[17 + 2 * n];
                float f2 = in[26 + 2 * n], f3 = in[27 + 2 * n];
                float p = f0 * w0 + f1 * w1 + f2 * w2 + f3 * w3 + b;
                float h = p > 0.f ? p : 0.01f * p;
                g0 += h * f0; g1 += h * f1; g2 += h * f2; g3 += h * f3; g4 += h;
            }
            int o = 320 + j * 5;
            putHL(Hh, Hl, o + 0, g0); putHL(Hh, Hl, o + 1, g1);
            putHL(Hh, Hl, o + 2, g2); putHL(Hh, Hl, o + 3, g3);
            putHL(Hh, Hl, o + 4, g4);
        }
        // landmark branch (d=2, 3 entities): cols 4..9
        {
            float w0 = lW1[j], w1 = lW1[64 + j];
            float b = lb1[j];
            float g0 = 0.f, g1 = 0.f, g2 = 0.f;
            #pragma unroll
            for (int n = 0; n < 3; ++n) {
                float f0 = in[4 + 2 * n], f1 = in[5 + 2 * n];
                float p = f0 * w0 + f1 * w1 + b;
                float h = p > 0.f ? p : 0.01f * p;
                g0 += h * f0; g1 += h * f1; g2 += h;
            }
            int o = 640 + j * 3;
            putHL(Hh, Hl, o + 0, g0); putHL(Hh, Hl, o + 1, g1); putHL(Hh, Hl, o + 2, g2);
        }
    }
    // Bias-coefficient K rows 832..847 + zero pad 848..863: 32*32 = 1024 items.
    #pragma unroll
    for (int it = 0; it < 4; ++it) {
        int idx = t + it * 256;
        int s = idx >> 5, kk = idx & 31;
        const float* in = inp + (size_t)(s0 + s) * 30;
        float v;
        switch (kk) {
            case 0:  v = in[10] + in[12] + in[14]; break;
            case 1:  v = in[11] + in[13] + in[15]; break;
            case 2:  v = in[20] + in[22] + in[24]; break;
            case 3:  v = in[21] + in[23] + in[25]; break;
            case 4:  v = 3.0f; break;
            case 5:  v = in[16] + in[18]; break;
            case 6:  v = in[17] + in[19]; break;
            case 7:  v = in[26] + in[28]; break;
            case 8:  v = in[27] + in[29]; break;
            case 9:  v = 2.0f; break;
            case 10: v = in[4] + in[6] + in[8]; break;
            case 11: v = in[5] + in[7] + in[9]; break;
            case 12: v = 3.0f; break;
            default: v = 0.0f; break;   // includes zero pad K rows
        }
        putHL(Ghi + s * KS, Glo + s * KS, 832 + kk, v);
    }
    __syncthreads();

    // ------------------ Phase 2: split-f16 WMMA GEMM, K = 864 ------------------
    const int lane = t & 31;
    const int wv   = t >> 5;          // wave 0..7, owns 32 output cols x 32 rows
    const int lo   = lane & 15;
    const int hi   = lane >> 4;       // lane-half: selects K sub-range per ISA layout
    const int n0   = wv * 32;

    v8f acc00 = {};   // rows 0..15,  cols n0..n0+15
    v8f acc01 = {};   // rows 0..15,  cols n0+16..n0+31
    v8f acc10 = {};   // rows 16..31, cols n0..n0+15
    v8f acc11 = {};   // rows 16..31, cols n0+16..n0+31
    const _Float16* A0h = Ghi + lo * KS + 8 * hi;          // A rows 0..15
    const _Float16* A0l = Glo + lo * KS + 8 * hi;
    const _Float16* A1h = Ghi + (16 + lo) * KS + 8 * hi;   // A rows 16..31
    const _Float16* A1l = Glo + (16 + lo) * KS + 8 * hi;
    const _Float16* B0h = W2h + (size_t)(n0 + lo) * KPS + 8 * hi;
    const _Float16* B0l = W2l + (size_t)(n0 + lo) * KPS + 8 * hi;
    const _Float16* B1h = W2h + (size_t)(n0 + 16 + lo) * KPS + 8 * hi;
    const _Float16* B1l = W2l + (size_t)(n0 + 16 + lo) * KPS + 8 * hi;

    #pragma unroll 3
    for (int kb = 0; kb < KTOT; kb += 32) {
        v16h a0h = ldA(A0h + kb), a0l = ldA(A0l + kb);
        v16h a1h = ldA(A1h + kb), a1l = ldA(A1l + kb);
        v16h b0h = ldB(B0h + kb), b0l = ldB(B0l + kb);
        v16h b1h = ldB(B1h + kb), b1l = ldB(B1l + kb);
        // hi*hi
        acc00 = __builtin_amdgcn_wmma_f32_16x16x32_f16(false, a0h, false, b0h, (short)0, acc00, false, false);
        acc01 = __builtin_amdgcn_wmma_f32_16x16x32_f16(false, a0h, false, b1h, (short)0, acc01, false, false);
        acc10 = __builtin_amdgcn_wmma_f32_16x16x32_f16(false, a1h, false, b0h, (short)0, acc10, false, false);
        acc11 = __builtin_amdgcn_wmma_f32_16x16x32_f16(false, a1h, false, b1h, (short)0, acc11, false, false);
        // hi*lo
        acc00 = __builtin_amdgcn_wmma_f32_16x16x32_f16(false, a0h, false, b0l, (short)0, acc00, false, false);
        acc01 = __builtin_amdgcn_wmma_f32_16x16x32_f16(false, a0h, false, b1l, (short)0, acc01, false, false);
        acc10 = __builtin_amdgcn_wmma_f32_16x16x32_f16(false, a1h, false, b0l, (short)0, acc10, false, false);
        acc11 = __builtin_amdgcn_wmma_f32_16x16x32_f16(false, a1h, false, b1l, (short)0, acc11, false, false);
        // lo*hi
        acc00 = __builtin_amdgcn_wmma_f32_16x16x32_f16(false, a0l, false, b0h, (short)0, acc00, false, false);
        acc01 = __builtin_amdgcn_wmma_f32_16x16x32_f16(false, a0l, false, b1h, (short)0, acc01, false, false);
        acc10 = __builtin_amdgcn_wmma_f32_16x16x32_f16(false, a1l, false, b0h, (short)0, acc10, false, false);
        acc11 = __builtin_amdgcn_wmma_f32_16x16x32_f16(false, a1l, false, b1h, (short)0, acc11, false, false);
    }
    __syncthreads();   // everyone done reading G before overlay reuse

    // Spill C tiles (VGPR r: M=r+8*hi, N=lane%16) to tmpS[MT][260].
    #pragma unroll
    for (int r = 0; r < 8; ++r) {
        int m = r + hi * 8;
        tmpS[m * 260 + n0 + lo]             = acc00[r];
        tmpS[m * 260 + n0 + 16 + lo]        = acc01[r];
        tmpS[(16 + m) * 260 + n0 + lo]      = acc10[r];
        tmpS[(16 + m) * 260 + n0 + 16 + lo] = acc11[r];
    }
    __syncthreads();

    // ------------------ Phase 3: epilogue ------------------
    // softmax(merger_w) over the 4 heads, per hid column (batch-invariant).
    if (t < 64) {
        float w0 = mw[t], w1 = mw[64 + t], w2 = mw[128 + t], w3 = mw[192 + t];
        float mx = fmaxf(fmaxf(w0, w1), fmaxf(w2, w3));
        float e0 = __expf(w0 - mx), e1 = __expf(w1 - mx);
        float e2 = __expf(w2 - mx), e3 = __expf(w3 - mx);
        float inv = 1.0f / (e0 + e1 + e2 + e3);
        swS[t] = e0 * inv; swS[64 + t] = e1 * inv;
        swS[128 + t] = e2 * inv; swS[192 + t] = e3 * inv;
    }
    __syncthreads();

    // merged + self branch + relu -> out66 (64 features + 2 actions)
    #pragma unroll
    for (int it = 0; it < 8; ++it) {
        int item = t + it * 256;
        int s = item >> 6;
        int hid = item & 63;
        const float* tp = tmpS + s * 260;
        float mg = swS[hid] * tp[hid] + swS[64 + hid] * tp[64 + hid]
                 + swS[128 + hid] * tp[128 + hid] + swS[192 + hid] * tp[192 + hid];
        const float* in = inp + (size_t)(s0 + s) * 30;
        float so = in[0] * selfW[hid] + in[1] * selfW[64 + hid]
                 + in[2] * selfW[128 + hid] + in[3] * selfW[192 + hid] + selfb[hid];
        float v = so + mg;
        o66S[s * 68 + hid] = v > 0.f ? v : 0.f;
        if (hid == 0) {
            o66S[s * 68 + 64] = act[(size_t)(s0 + s) * 2];
            o66S[s * 68 + 65] = act[(size_t)(s0 + s) * 2 + 1];
        }
    }
    __syncthreads();

    // fc1 (66 -> 64) + relu; write x to output and stash for fc2.
    #pragma unroll
    for (int it = 0; it < 8; ++it) {
        int item = t + it * 256;
        int s = item >> 6;
        int h2 = item & 63;
        float accv = f1b[h2];
        const float* o = o66S + s * 68;
        #pragma unroll 6
        for (int i = 0; i < 66; ++i) accv += o[i] * f1W[i * 64 + h2];
        float x = accv > 0.f ? accv : 0.f;
        xS[s * 64 + h2] = x;
        outq[(size_t)BSZ + (size_t)(s0 + s) * 64 + h2] = x;
    }
    __syncthreads();

    // fc2 (64 -> 1): q
    if (t < MT) {
        float q = f2b[0];
        const float* x = xS + t * 64;
        #pragma unroll 8
        for (int i = 0; i < 64; ++i) q += x[i] * f2W[i];
        outq[s0 + t] = q;
    }
}

extern "C" void kernel_launch(void* const* d_in, const int* in_sizes, int n_in,
                              void* d_out, int out_size, void* d_ws, size_t ws_size,
                              hipStream_t stream) {
    const float* inputs = (const float*)d_in[0];
    // d_in[1] = hidden_state (unused by the reference)
    const float* actions = (const float*)d_in[2];
    const float* selfW = (const float*)d_in[3];
    const float* selfb = (const float*)d_in[4];
    const float* aW1   = (const float*)d_in[5];
    const float* ab1   = (const float*)d_in[6];
    const float* aW2   = (const float*)d_in[7];
    const float* ab2   = (const float*)d_in[8];
    const float* dW1   = (const float*)d_in[9];
    const float* db1   = (const float*)d_in[10];
    const float* dW2   = (const float*)d_in[11];
    const float* db2   = (const float*)d_in[12];
    const float* lW1   = (const float*)d_in[13];
    const float* lb1   = (const float*)d_in[14];
    const float* lW2   = (const float*)d_in[15];
    const float* lb2   = (const float*)d_in[16];
    const float* mw    = (const float*)d_in[17];
    const float* f1W   = (const float*)d_in[18];
    const float* f1b   = (const float*)d_in[19];
    const float* f2W   = (const float*)d_in[20];
    const float* f2b   = (const float*)d_in[21];

    _Float16* W2h = (_Float16*)d_ws;           // 256*872*2 B
    _Float16* W2l = W2h + 256 * KPS;           // 256*872*2 B  (total ~0.89 MB)

    pack_w2<<<256, 256, 0, stream>>>(aW2, ab2, dW2, db2, lW2, lb2, W2h, W2l);
    hpn_main<<<BSZ / MT, 256, 0, stream>>>(inputs, actions, selfW, selfb,
                                           aW1, ab1, dW1, db1, lW1, lb1,
                                           mw, f1W, f1b, f2W, f2b,
                                           W2h, W2l, (float*)d_out);
}
